// GeometricTransformer_42090679501113
// MI455X (gfx1250) — matrix operand order repaired and checked by reading, not compiled, not run on hardware
//
#include <hip/hip_runtime.h>
#include <math.h>

// MI455X / gfx1250, wave32. D = A(16x32 f16) x B(32x16 f16) + C(16x16 f32)
typedef __attribute__((ext_vector_type(16))) _Float16 v16h;
typedef __attribute__((ext_vector_type(8)))  _Float16 v8h;
typedef __attribute__((ext_vector_type(8)))  float    v8f;

#define HIDDEN 128
#define NPTS   512

// constants from reference
// 1/SIGMA_D = 5.0 ; FACTOR_A = 180/(15*pi) ; div[f] = exp(-f*ln(10000)/64)
__device__ __forceinline__ v8f splat8(float v) {
  v8f r;
#pragma unroll
  for (int i = 0; i < 8; ++i) r[i] = v;
  return r;
}

// ---------------- W -> f16 conversion (weights stay hot in L2/workspace) ---
__global__ void convert_w_kernel(const float* __restrict__ Wd,
                                 const float* __restrict__ Wa,
                                 _Float16* __restrict__ hWd,
                                 _Float16* __restrict__ hWa) {
  int i = blockIdx.x * blockDim.x + threadIdx.x;
  if (i < HIDDEN * HIDDEN) {
    hWd[i] = (_Float16)Wd[i];
    hWa[i] = (_Float16)Wa[i];
  }
}

// ---------------- 3-NN reference vectors ----------------------------------
// Stable 4-smallest scan (strict < insert keeps ascending-index tie order,
// matching jax.lax.top_k). Slot 0 is self (dist 0); ref vectors from 1..3.
__global__ void knn_refvec_kernel(const float* __restrict__ pts,
                                  float* __restrict__ refv) {
  const int n = blockIdx.x * blockDim.x + threadIdx.x;
  if (n >= NPTS) return;
  const float px = pts[n * 3 + 0], py = pts[n * 3 + 1], pz = pts[n * 3 + 2];
  float b0 = 3.0e38f, b1 = 3.0e38f, b2 = 3.0e38f, b3 = 3.0e38f;
  int i0 = 0, i1 = 0, i2 = 0, i3 = 0;
  for (int mm = 0; mm < NPTS; ++mm) {
    const float dx = pts[mm * 3 + 0] - px;
    const float dy = pts[mm * 3 + 1] - py;
    const float dz = pts[mm * 3 + 2] - pz;
    const float d2 = dx * dx + dy * dy + dz * dz;
    if (d2 < b3) {
      if (d2 < b0)      { b3=b2;i3=i2; b2=b1;i2=i1; b1=b0;i1=i0; b0=d2;i0=mm; }
      else if (d2 < b1) { b3=b2;i3=i2; b2=b1;i2=i1; b1=d2;i1=mm; }
      else if (d2 < b2) { b3=b2;i3=i2; b2=d2;i2=mm; }
      else              { b3=d2;i3=mm; }
    }
  }
  const int nb[3] = { i1, i2, i3 };
#pragma unroll
  for (int k = 0; k < 3; ++k) {
    const int q = nb[k];
    refv[(n * 3 + k) * 3 + 0] = pts[q * 3 + 0] - px;
    refv[(n * 3 + k) * 3 + 1] = pts[q * 3 + 1] - py;
    refv[(n * 3 + k) * 3 + 2] = pts[q * 3 + 2] - pz;
  }
}

// ---------------- fused embedding + WMMA projection ------------------------
// WG = 128 threads = 4 waves; each wave owns 16 (n,m) pairs (fixed n).
// Per-wave LDS: 4 embedding tiles (d, a0, a1, a2), each 16x128 f16.
__global__ __launch_bounds__(128) void geo_main_kernel(
    const float* __restrict__ pts,
    const _Float16* __restrict__ hWd,
    const _Float16* __restrict__ hWa,
    const float* __restrict__ bdv,
    const float* __restrict__ bav,
    const float* __restrict__ refv,
    float* __restrict__ out) {
  __shared__ __align__(16) _Float16 semb[4][4][16][HIDDEN];  // 64 KB

  const int tid  = threadIdx.x;
  const int wave = tid >> 5;
  const int lane = tid & 31;
  const int row  = lane & 15;   // A-matrix M index owned by this lane
  const int hlf  = lane >> 4;   // half-wave select

  const int wg    = blockIdx.x;           // 512 n * 8 m-tiles = 4096 WGs
  const int n     = wg >> 3;
  const int mbase = (wg & 7) * 64 + wave * 16;
  const int m     = mbase + row;

  // ---- per-pair geometric indices (both half-waves compute same row) ----
  const float pnx = pts[n * 3 + 0], pny = pts[n * 3 + 1], pnz = pts[n * 3 + 2];
  const float ax = pts[m * 3 + 0] - pnx;
  const float ay = pts[m * 3 + 1] - pny;
  const float az = pts[m * 3 + 2] - pnz;

  float xv[4];
  xv[0] = sqrtf(ax * ax + ay * ay + az * az) * 5.0f;  // dist / SIGMA_D
#pragma unroll
  for (int k = 0; k < 3; ++k) {
    const float rx = refv[(n * 3 + k) * 3 + 0];
    const float ry = refv[(n * 3 + k) * 3 + 1];
    const float rz = refv[(n * 3 + k) * 3 + 2];
    const float cx = ry * az - rz * ay;
    const float cy = rz * ax - rx * az;
    const float cz = rx * ay - ry * ax;
    const float sv = sqrtf(cx * cx + cy * cy + cz * cz);
    const float cv = rx * ax + ry * ay + rz * az;
    xv[1 + k] = atan2f(sv, cv) * 3.8197186342054885f;  // FACTOR_A
  }

  // ---- sinusoidal embeddings -> LDS f16 tiles ----
  // lane covers freqs f = hlf*32 .. +31 of its row: sin at col f, cos at 64+f
  const int fbase = hlf * 32;
#pragma unroll
  for (int slot = 0; slot < 4; ++slot) {
    const float x = xv[slot];
    _Float16* dst = &semb[wave][slot][row][0];
    for (int i = 0; i < 32; ++i) {
      const int f = fbase + i;
      const float w = x * __expf(-0.14391156698393129f * (float)f);
      float s, c;
      __sincosf(w, &s, &c);
      dst[f]      = (_Float16)s;
      dst[64 + f] = (_Float16)c;
    }
  }
  __syncthreads();

  // ---- WMMA: [16x128] emb @ [128x16] W^T per 16-col output tile ----
  const int kbA = hlf * 16;   // A per-lane K base within a 32-wide chunk
  const int kbB = hlf * 16;   // B per-lane K base within a 32-wide chunk
  const int col = lane & 15;  // C/D column owned by this lane

  for (int ct = 0; ct < 8; ++ct) {
    const int ocol = ct * 16 + col;                  // output col == W row
    const _Float16* wdr = hWd + ocol * HIDDEN;
    const _Float16* war = hWa + ocol * HIDDEN;

    // distance branch
    v8f accD = splat8(0.0f);
#pragma unroll
    for (int c = 0; c < 4; ++c) {
      const v8h* ap = (const v8h*)&semb[wave][0][row][c * 32 + kbA];
      v16h a = __builtin_shufflevector(ap[0], ap[1],
               0, 1, 2, 3, 4, 5, 6, 7, 8, 9, 10, 11, 12, 13, 14, 15);
      v16h b = *(const v16h*)(wdr + c * 32 + kbB);
      accD = __builtin_amdgcn_wmma_f32_16x16x32_f16(
          false, a, false, b, (short)0, accD, false, false);
    }

    // angle branch: max over k of (emb_k @ Wa^T); bias added after max
    v8f accA = splat8(-3.0e38f);
#pragma unroll
    for (int k = 0; k < 3; ++k) {
      v8f t = splat8(0.0f);
#pragma unroll
      for (int c = 0; c < 4; ++c) {
        const v8h* ap = (const v8h*)&semb[wave][1 + k][row][c * 32 + kbA];
        v16h a = __builtin_shufflevector(ap[0], ap[1],
                 0, 1, 2, 3, 4, 5, 6, 7, 8, 9, 10, 11, 12, 13, 14, 15);
        v16h b = *(const v16h*)(war + c * 32 + kbB);
        t = __builtin_amdgcn_wmma_f32_16x16x32_f16(
            false, a, false, b, (short)0, t, false, false);
      }
#pragma unroll
      for (int i = 0; i < 8; ++i) accA[i] = fmaxf(accA[i], t[i]);
    }

    const float bsum = bdv[ocol] + bav[ocol];
#pragma unroll
    for (int v = 0; v < 8; ++v) {
      const int mm = mbase + v + hlf * 8;  // C layout: M = vgpr + 8*halfwave
      out[(size_t)(n * NPTS + mm) * HIDDEN + ocol] = accD[v] + accA[v] + bsum;
    }
  }
}

// ---------------- host-side launch -----------------------------------------
extern "C" void kernel_launch(void* const* d_in, const int* in_sizes, int n_in,
                              void* d_out, int out_size, void* d_ws,
                              size_t ws_size, hipStream_t stream) {
  (void)in_sizes; (void)n_in; (void)out_size; (void)ws_size;
  const float* pts = (const float*)d_in[0];
  const float* Wd  = (const float*)d_in[1];
  const float* bd  = (const float*)d_in[2];
  const float* Wa  = (const float*)d_in[3];
  const float* ba  = (const float*)d_in[4];
  float* out = (float*)d_out;

  // workspace: [hWd f16 32KB][hWa f16 32KB][refv f32 18KB]
  _Float16* hWd = (_Float16*)d_ws;
  _Float16* hWa = hWd + HIDDEN * HIDDEN;
  float* refv = (float*)((char*)d_ws + 2 * HIDDEN * HIDDEN * sizeof(_Float16));

  convert_w_kernel<<<(HIDDEN * HIDDEN + 255) / 256, 256, 0, stream>>>(
      Wd, Wa, hWd, hWa);
  knn_refvec_kernel<<<(NPTS + 255) / 256, 256, 0, stream>>>(pts, refv);
  geo_main_kernel<<<NPTS * 8, 128, 0, stream>>>(
      pts, hWd, hWa, bd, ba, refv, out);
}
